// Attention_10488310137333
// MI455X (gfx1250) — compile-verified, matrix operand order checked by reading, not run on hardware
//
#include <hip/hip_runtime.h>
#include <hip/hip_bf16.h>
#include <stdint.h>

// ---------------------------------------------------------------------------
// Types for CDNA5 WMMA (wave32)
// ---------------------------------------------------------------------------
typedef __attribute__((ext_vector_type(16))) __bf16 v16bf;
typedef __attribute__((ext_vector_type(8)))  float  v8f;

#define WMMA_BF16(A, B, C) \
    __builtin_amdgcn_wmma_f32_16x16x32_bf16(false, (A), false, (B), (short)0, (C), false, false)

// ---------------------------------------------------------------------------
// Async global->LDS (CDNA5 GLOBAL_LOAD_ASYNC_TO_LDS_B128, ASYNCcnt-tracked).
// Probe-confirmed signature: (int4 addrspace(1)*, int4 addrspace(3)*, imm, imm)
// ---------------------------------------------------------------------------
#if defined(__HIP_DEVICE_COMPILE__) && __has_builtin(__builtin_amdgcn_global_load_async_to_lds_b128)
#define HAVE_ASYNC_LDS 1
typedef int v4i __attribute__((ext_vector_type(4)));
typedef __attribute__((address_space(1))) v4i as1_v4i;
typedef __attribute__((address_space(3))) v4i as3_v4i;
__device__ __forceinline__ void async_ld_b128(const void* g, void* l) {
    __builtin_amdgcn_global_load_async_to_lds_b128((as1_v4i*)g, (as3_v4i*)l, 0, 0);
}
__device__ __forceinline__ void wait_async0() {
#if __has_builtin(__builtin_amdgcn_s_wait_asynccnt)
    __builtin_amdgcn_s_wait_asynccnt(0);
#else
    asm volatile("s_wait_asynccnt 0" ::: "memory");
#endif
}
#else
#define HAVE_ASYNC_LDS 0
#endif

// ---------------------------------------------------------------------------
// bf16 helpers (round-to-nearest-even)
// ---------------------------------------------------------------------------
__device__ __forceinline__ unsigned short f2bf(float x) {
    union { float f; unsigned u; } v; v.f = x;
    unsigned r = v.u + 0x7FFFu + ((v.u >> 16) & 1u);
    return (unsigned short)(r >> 16);
}
__device__ __forceinline__ float bf2f(unsigned short b) {
    union { unsigned u; float f; } v; v.u = ((unsigned)b) << 16;
    return v.f;
}

// Build a 16x32 bf16 A/B fragment from two 16-byte chunks (ISA layout:
// lane L holds 8 contiguous K at (L/16)*8 and 8 more at 16+(L/16)*8).
__device__ __forceinline__ v16bf make_frag(const unsigned short* p0, const unsigned short* p1) {
    v16bf f;
    ((uint4*)&f)[0] = *(const uint4*)p0;
    ((uint4*)&f)[1] = *(const uint4*)p1;
    return f;
}

// ---------------------------------------------------------------------------
// fp32 -> bf16 conversion (4 elements per thread)
// ---------------------------------------------------------------------------
__global__ __launch_bounds__(256) void cvt_f32_bf16(const float* __restrict__ in,
                                                    unsigned short* __restrict__ out,
                                                    size_t n) {
    size_t i = ((size_t)blockIdx.x * 256 + threadIdx.x) * 4;
    if (i + 3 < n) {
        float4 v = *(const float4*)(in + i);
        out[i + 0] = f2bf(v.x);
        out[i + 1] = f2bf(v.y);
        out[i + 2] = f2bf(v.z);
        out[i + 3] = f2bf(v.w);
    }
}

// ---------------------------------------------------------------------------
// K-major bf16 GEMM:  C[m,n] = sum_k A[m*K+k] * B[n*K+k]
// 128x128 block tile, 8 waves in 2(M)x4(N) grid, each wave 64x32 (8 WMMAs/step)
// Async path: double-buffered LDS, next tile in flight during WMMA compute.
// ---------------------------------------------------------------------------
template <bool F32OUT>
__global__ __launch_bounds__(256) void gemm_bf16_kmajor(const unsigned short* __restrict__ A,
                                                        const unsigned short* __restrict__ Bw,
                                                        void* __restrict__ C,
                                                        int M, int N, int K) {
    __shared__ unsigned short ldsA[2][128 * 40];   // 32-K tiles, row stride 40 (16B rows)
    __shared__ unsigned short ldsB[2][128 * 40];

    const int tid  = threadIdx.x;
    const int wave = tid >> 5;
    const int lane = tid & 31;
    const int g    = lane >> 4;      // half-wave group
    const int l16  = lane & 15;

    const int bm = blockIdx.y * 128;
    const int bn = blockIdx.x * 128;
    const int wm = (wave >> 2) * 64; // wave M offset inside block tile
    const int wn = (wave & 3) * 32;  // wave N offset inside block tile

    v8f acc[4][2];
#pragma unroll
    for (int i = 0; i < 4; ++i)
#pragma unroll
        for (int j = 0; j < 2; ++j)
#pragma unroll
            for (int r = 0; r < 8; ++r) acc[i][j][r] = 0.0f;

    const int row_ld = tid >> 1;          // 0..127
    const int half   = (tid & 1) * 16;    // 0 or 16 (K elements)

#if HAVE_ASYNC_LDS
    // ---- async double-buffered pipeline ----
    {
        const unsigned short* ga = A  + (size_t)(bm + row_ld) * K + half;
        const unsigned short* gb = Bw + (size_t)(bn + row_ld) * K + half;
        async_ld_b128(ga,     &ldsA[0][row_ld * 40 + half]);
        async_ld_b128(ga + 8, &ldsA[0][row_ld * 40 + half + 8]);
        async_ld_b128(gb,     &ldsB[0][row_ld * 40 + half]);
        async_ld_b128(gb + 8, &ldsB[0][row_ld * 40 + half + 8]);
    }
    wait_async0();
    __syncthreads();

    for (int k0 = 0; k0 < K; k0 += 32) {
        const int buf = (k0 >> 5) & 1;
        if (k0 + 32 < K) {   // stage next tile into the other buffer (async)
            const unsigned short* ga = A  + (size_t)(bm + row_ld) * K + k0 + 32 + half;
            const unsigned short* gb = Bw + (size_t)(bn + row_ld) * K + k0 + 32 + half;
            async_ld_b128(ga,     &ldsA[buf ^ 1][row_ld * 40 + half]);
            async_ld_b128(ga + 8, &ldsA[buf ^ 1][row_ld * 40 + half + 8]);
            async_ld_b128(gb,     &ldsB[buf ^ 1][row_ld * 40 + half]);
            async_ld_b128(gb + 8, &ldsB[buf ^ 1][row_ld * 40 + half + 8]);
        }

        v16bf afr[4];
#pragma unroll
        for (int mt = 0; mt < 4; ++mt) {
            const int row = wm + mt * 16 + l16;
            afr[mt] = make_frag(&ldsA[buf][row * 40 + g * 8], &ldsA[buf][row * 40 + 16 + g * 8]);
        }
#pragma unroll
        for (int nt = 0; nt < 2; ++nt) {
            const int row = wn + nt * 16 + l16;
            v16bf bfr = make_frag(&ldsB[buf][row * 40 + g * 8], &ldsB[buf][row * 40 + 16 + g * 8]);
#pragma unroll
            for (int mt = 0; mt < 4; ++mt)
                acc[mt][nt] = WMMA_BF16(afr[mt], bfr, acc[mt][nt]);
        }

        wait_async0();     // next tile landed in LDS
        __syncthreads();   // all waves done reading current buffer
    }
#else
    // ---- fallback: global->VGPR->LDS staging (single buffer) ----
    for (int k0 = 0; k0 < K; k0 += 32) {
        const unsigned short* ga = A  + (size_t)(bm + row_ld) * K + k0 + half;
        const unsigned short* gb = Bw + (size_t)(bn + row_ld) * K + k0 + half;
        uint4 a0 = ((const uint4*)ga)[0];
        uint4 a1 = ((const uint4*)ga)[1];
        uint4 b0 = ((const uint4*)gb)[0];
        uint4 b1 = ((const uint4*)gb)[1];
        if (k0 + 32 < K) {
            __builtin_prefetch(ga + 32, 0, 3);
            __builtin_prefetch(gb + 32, 0, 3);
        }

        __syncthreads();
        *(uint4*)&ldsA[0][row_ld * 40 + half    ] = a0;
        *(uint4*)&ldsA[0][row_ld * 40 + half + 8] = a1;
        *(uint4*)&ldsB[0][row_ld * 40 + half    ] = b0;
        *(uint4*)&ldsB[0][row_ld * 40 + half + 8] = b1;
        __syncthreads();

        v16bf afr[4];
#pragma unroll
        for (int mt = 0; mt < 4; ++mt) {
            const int row = wm + mt * 16 + l16;
            afr[mt] = make_frag(&ldsA[0][row * 40 + g * 8], &ldsA[0][row * 40 + 16 + g * 8]);
        }
#pragma unroll
        for (int nt = 0; nt < 2; ++nt) {
            const int row = wn + nt * 16 + l16;
            v16bf bfr = make_frag(&ldsB[0][row * 40 + g * 8], &ldsB[0][row * 40 + 16 + g * 8]);
#pragma unroll
            for (int mt = 0; mt < 4; ++mt)
                acc[mt][nt] = WMMA_BF16(afr[mt], bfr, acc[mt][nt]);
        }
    }
#endif

    // Epilogue: C-fragment layout — lane (g,l16) holds rows g*8+r, col l16.
#pragma unroll
    for (int mt = 0; mt < 4; ++mt)
#pragma unroll
        for (int nt = 0; nt < 2; ++nt)
#pragma unroll
            for (int r = 0; r < 8; ++r) {
                const int gm = bm + wm + mt * 16 + g * 8 + r;
                const int gn = bn + wn + nt * 16 + l16;
                const float v = acc[mt][nt][r];
                if (F32OUT) ((float*)C)[(size_t)gm * N + gn] = v;
                else        ((unsigned short*)C)[(size_t)gm * N + gn] = f2bf(v);
            }
}

// ---------------------------------------------------------------------------
// RoPE + scatter: qkv[B*S, 3*4096] (bf16) -> Qr/Kr/Vr [B, 32, S, 128] (bf16)
// ---------------------------------------------------------------------------
__global__ __launch_bounds__(256) void rope_scatter(const unsigned short* __restrict__ qkv,
                                                    unsigned short* __restrict__ Qr,
                                                    unsigned short* __restrict__ Kr,
                                                    unsigned short* __restrict__ Vr) {
    const unsigned t = blockIdx.x * 256 + threadIdx.x;   // < 2*2048*32*64
    const int j = t & 63;
    const int h = (t >> 6) & 31;
    const int s = (t >> 11) & 2047;
    const int b = t >> 22;

    const size_t rowbase = (size_t)(b * 2048 + s) * 12288 + (size_t)h * 128;
    const float q1 = bf2f(qkv[rowbase + j]);
    const float q2 = bf2f(qkv[rowbase + j + 64]);
    const float k1 = bf2f(qkv[rowbase + 4096 + j]);
    const float k2 = bf2f(qkv[rowbase + 4096 + j + 64]);

    const float inv = __powf(10000.0f, -(float)j * (1.0f / 64.0f));
    const float th  = (float)s * inv;
    float sn, cs;
    __sincosf(th, &sn, &cs);

    const size_t obase = (((size_t)(b * 32 + h)) * 2048 + (size_t)s) * 128;
    Qr[obase + j]      = f2bf(q1 * cs - q2 * sn);
    Qr[obase + j + 64] = f2bf(q2 * cs + q1 * sn);
    Kr[obase + j]      = f2bf(k1 * cs - k2 * sn);
    Kr[obase + j + 64] = f2bf(k2 * cs + k1 * sn);
    Vr[obase + j]      = qkv[rowbase + 8192 + j];
    Vr[obase + j + 64] = qkv[rowbase + 8192 + j + 64];
}

// ---------------------------------------------------------------------------
// Flash attention (causal). One workgroup = 8 waves = 128 q rows of one head.
// K tile staged via async global->LDS (overlapping the VGPR-based V transpose
// stores); V stored TRANSPOSED so PV B-fragments are two aligned 16B chunks.
// ---------------------------------------------------------------------------
__global__ __launch_bounds__(256) void flash_attn(const unsigned short* __restrict__ Qr,
                                                  const unsigned short* __restrict__ Kr,
                                                  const unsigned short* __restrict__ Vr,
                                                  unsigned short* __restrict__ attn) {
    __shared__ unsigned short ldsK[32 * 136];    // [key][d]   stride 136 (16B-mult, skewed)
    __shared__ unsigned short ldsVT[128 * 40];   // [d][key]   stride 40
    __shared__ unsigned short ldsP[8][16 * 40];  // per-wave 16x32 P tile

    const int bh   = blockIdx.x;          // b*32 + h
    const int qblk = blockIdx.y;          // 0..15
    const int b    = bh >> 5;
    const int h    = bh & 31;
    const int tid  = threadIdx.x;
    const int wave = tid >> 5;
    const int lane = tid & 31;
    const int g    = lane >> 4;
    const int l16  = lane & 15;
    const int q0   = qblk * 128 + wave * 16;

    const size_t head_off = (size_t)bh * 2048 * 128;
    const unsigned short* Qh = Qr + head_off;
    const unsigned short* Kh = Kr + head_off;
    const unsigned short* Vh = Vr + head_off;

    // Q fragments for all 4 K-dim chunks (d = 0..127), kept in VGPRs
    v16bf qa[4];
#pragma unroll
    for (int c = 0; c < 4; ++c) {
        const unsigned short* qp = &Qh[(size_t)(q0 + l16) * 128 + c * 32];
        qa[c] = make_frag(qp + g * 8, qp + 16 + g * 8);
    }

    v8f o[8];
    float mi[8], li[8];
#pragma unroll
    for (int d = 0; d < 8; ++d) {
#pragma unroll
        for (int r = 0; r < 8; ++r) o[d][r] = 0.0f;
        mi[d] = -3.0e38f;
        li[d] = 0.0f;
    }

    unsigned short* P = &ldsP[wave][0];
    const float scale = 0.08838834764831845f;   // 1/sqrt(128)

    // cooperative tile-load indices: 256 threads x 32B = one 32x128 bf16 tile
    const int krow = tid >> 3;            // 0..31 (key row in block)
    const int dcol = (tid & 7) * 16;      // 0,16,...,112

    const int kend_wg = qblk * 128 + 128; // WG-wide causal bound
    const int kend_w  = q0 + 16;          // this wave's causal bound

    for (int kk = 0; kk < kend_wg; kk += 32) {
        // ---- cooperative staging of K (async) and V^T (via VGPRs) ----
        const unsigned short* kg = &Kh[(size_t)(kk + krow) * 128 + dcol];
        const unsigned short* vg = &Vh[(size_t)(kk + krow) * 128 + dcol];
        uint4 va = ((const uint4*)vg)[0];
        uint4 vb = ((const uint4*)vg)[1];
        if (kk + 32 < kend_wg) {                 // prefetch next V tile
            __builtin_prefetch(vg + 32 * 128, 0, 3);
        }

        __syncthreads();
#if HAVE_ASYNC_LDS
        async_ld_b128(kg,     &ldsK[krow * 136 + dcol]);
        async_ld_b128(kg + 8, &ldsK[krow * 136 + dcol + 8]);
#else
        {
            uint4 ka = ((const uint4*)kg)[0];
            uint4 kb = ((const uint4*)kg)[1];
            *(uint4*)&ldsK[krow * 136 + dcol    ] = ka;
            *(uint4*)&ldsK[krow * 136 + dcol + 8] = kb;
        }
#endif
        {
            const unsigned short* ve = (const unsigned short*)&va;
#pragma unroll
            for (int e = 0; e < 8; ++e)
                ldsVT[(dcol + e) * 40 + krow] = ve[e];
            const unsigned short* vf = (const unsigned short*)&vb;
#pragma unroll
            for (int e = 0; e < 8; ++e)
                ldsVT[(dcol + 8 + e) * 40 + krow] = vf[e];
        }
#if HAVE_ASYNC_LDS
        wait_async0();
#endif
        __syncthreads();

        if (kk < kend_w) {
            // --- scores: two 16x16 tiles, K fragments from LDS ---
            v8f s[2];
#pragma unroll
            for (int j = 0; j < 2; ++j) {
#pragma unroll
                for (int r = 0; r < 8; ++r) s[j][r] = 0.0f;
                const int keyl = j * 16 + l16;
#pragma unroll
                for (int c = 0; c < 4; ++c) {
                    v16bf kf = make_frag(&ldsK[keyl * 136 + c * 32 + g * 8],
                                         &ldsK[keyl * 136 + c * 32 + 16 + g * 8]);
                    s[j] = WMMA_BF16(qa[c], kf, s[j]);
                }
            }
            // --- scale + causal mask ---
#pragma unroll
            for (int j = 0; j < 2; ++j)
#pragma unroll
                for (int r = 0; r < 8; ++r) {
                    const int row = q0 + g * 8 + r;
                    const int col = kk + j * 16 + l16;
                    float v = s[j][r] * scale;
                    if (col > row) v = -3.0e38f;
                    s[j][r] = v;
                }
            // --- online softmax (row reductions across 16-lane groups) ---
            float alpha[8];
#pragma unroll
            for (int r = 0; r < 8; ++r) {
                float mx = fmaxf(s[0][r], s[1][r]);
#pragma unroll
                for (int d = 1; d < 16; d <<= 1) mx = fmaxf(mx, __shfl_xor(mx, d, 32));
                const float mnew = fmaxf(mi[r], mx);
                alpha[r] = __expf(mi[r] - mnew);
                float rs = 0.0f;
#pragma unroll
                for (int j = 0; j < 2; ++j) {
                    const float p = __expf(s[j][r] - mnew);
                    s[j][r] = p;
                    rs += p;
                }
#pragma unroll
                for (int d = 1; d < 16; d <<= 1) rs += __shfl_xor(rs, d, 32);
                li[r] = li[r] * alpha[r] + rs;
                mi[r] = mnew;
            }
#pragma unroll
            for (int dc = 0; dc < 8; ++dc)
#pragma unroll
                for (int r = 0; r < 8; ++r) o[dc][r] *= alpha[r];

            // --- P (C layout) -> per-wave LDS -> A-fragment (in-order DS) ---
#pragma unroll
            for (int j = 0; j < 2; ++j)
#pragma unroll
                for (int r = 0; r < 8; ++r)
                    P[(g * 8 + r) * 40 + j * 16 + l16] = f2bf(s[j][r]);

            v16bf pfr = make_frag(&P[l16 * 40 + g * 8], &P[l16 * 40 + 16 + g * 8]);

            // --- PV: V^T fragments from LDS (aligned 16B chunks) ---
#pragma unroll
            for (int dc = 0; dc < 8; ++dc) {
                const int drow = dc * 16 + l16;
                v16bf vfr = make_frag(&ldsVT[drow * 40 + g * 8],
                                      &ldsVT[drow * 40 + 16 + g * 8]);
                o[dc] = WMMA_BF16(pfr, vfr, o[dc]);
            }
        }
    }

    // --- normalize + write [B, S, 4096] bf16 ---
#pragma unroll
    for (int r = 0; r < 8; ++r) {
        const float inv_l = 1.0f / li[r];
        const int row = q0 + g * 8 + r;
        const size_t base = ((size_t)(b * 2048) + row) * 4096 + (size_t)h * 128;
#pragma unroll
        for (int dc = 0; dc < 8; ++dc)
            attn[base + dc * 16 + l16] = f2bf(o[dc][r] * inv_l);
    }
}

// ---------------------------------------------------------------------------
// Launch
// ---------------------------------------------------------------------------
extern "C" void kernel_launch(void* const* d_in, const int* in_sizes, int n_in,
                              void* d_out, int out_size, void* d_ws, size_t ws_size,
                              hipStream_t stream) {
    const float* hs     = (const float*)d_in[0];   // [2,2048,4096]
    // d_in[1] attention_mask (causal, implemented analytically)
    // d_in[2] position_ids   (arange, implemented analytically)
    const float* W_pack = (const float*)d_in[3];   // [12288,4096]
    const float* W_o    = (const float*)d_in[4];   // [4096,4096]

    const size_t N_HS    = (size_t)2 * 2048 * 4096;         // 16,777,216
    const size_t N_WPACK = (size_t)12288 * 4096;            // 50,331,648
    const size_t N_WO    = (size_t)4096 * 4096;             // 16,777,216
    const size_t N_HEADS_BUF = (size_t)2 * 32 * 2048 * 128; // 16,777,216

    unsigned short* ws     = (unsigned short*)d_ws;
    unsigned short* hsb    = ws;                     // bf16 hidden
    unsigned short* wpackb = hsb    + N_HS;
    unsigned short* wob    = wpackb + N_WPACK;
    unsigned short* qkvb   = wob    + N_WO;          // [4096, 12288]
    unsigned short* Qrb    = qkvb   + N_WPACK;
    unsigned short* Krb    = Qrb    + N_HEADS_BUF;
    unsigned short* Vrb    = Krb    + N_HEADS_BUF;
    unsigned short* attnb  = Vrb    + N_HEADS_BUF;   // [4096, 4096] bf16

    // 1) fp32 -> bf16
    cvt_f32_bf16<<<(unsigned)(N_HS    / 1024), 256, 0, stream>>>(hs,     hsb,    N_HS);
    cvt_f32_bf16<<<(unsigned)(N_WPACK / 1024), 256, 0, stream>>>(W_pack, wpackb, N_WPACK);
    cvt_f32_bf16<<<(unsigned)(N_WO    / 1024), 256, 0, stream>>>(W_o,    wob,    N_WO);

    // 2) QKV projection: [4096,4096] x [12288,4096]^T -> [4096,12288]
    gemm_bf16_kmajor<false><<<dim3(12288 / 128, 4096 / 128), 256, 0, stream>>>(
        hsb, wpackb, qkvb, 4096, 12288, 4096);

    // 3) RoPE + per-head scatter
    rope_scatter<<<(unsigned)((size_t)2 * 2048 * 32 * 64 / 256), 256, 0, stream>>>(
        qkvb, Qrb, Krb, Vrb);

    // 4) causal flash attention
    flash_attn<<<dim3(64, 16), 256, 0, stream>>>(Qrb, Krb, Vrb, attnb);

    // 5) output projection: [4096,4096] x [4096,4096]^T -> fp32 d_out
    gemm_bf16_kmajor<true><<<dim3(4096 / 128, 4096 / 128), 256, 0, stream>>>(
        attnb, wob, d_out, 4096, 4096, 4096);
}
